// DistancePolicyValueHead_46969762349635
// MI455X (gfx1250) — matrix operand order repaired and checked by reading, not compiled, not run on hardware
//
#include <hip/hip_runtime.h>
#include <math.h>

typedef __attribute__((ext_vector_type(16))) __bf16 bf16x16;
typedef __attribute__((ext_vector_type(8)))  __bf16 bf16x8;
typedef __attribute__((ext_vector_type(8)))  float  f32x8;

#define EPSN 1e-8f

// ---------------- elementwise fp32 -> bf16 ----------------
__global__ __launch_bounds__(256)
void k_f32_to_bf16(const float* __restrict__ in, __bf16* __restrict__ out, int n) {
  int i = blockIdx.x * 256 + threadIdx.x;
  if (i < n) out[i] = (__bf16)in[i];
}

// ------------- transpose [R,C] fp32 -> [C,R] bf16 -------------
// R, C are multiples of 32. block (32,8), grid (C/32, R/32).
__global__ __launch_bounds__(256)
void k_transpose_bf16(const float* __restrict__ in, __bf16* __restrict__ out,
                      int R, int C) {
  __shared__ float tile[32][33];
  const int tx = threadIdx.x, ty = threadIdx.y;
  const int c0 = blockIdx.x * 32, r0 = blockIdx.y * 32;
#pragma unroll
  for (int j = 0; j < 32; j += 8)
    tile[ty + j][tx] = in[(size_t)(r0 + ty + j) * C + (c0 + tx)];
  __syncthreads();
#pragma unroll
  for (int j = 0; j < 32; j += 8)
    out[(size_t)(c0 + ty + j) * R + (r0 + tx)] = (__bf16)tile[tx][ty + j];
}

// ---------------- WMMA GEMM:  Out = epilogue(A[M,K] @ BT[N,K]^T) ----------------
// A row-major bf16, BT row-major bf16 (i.e. B transposed).
// Per-wave 64x64 tile (4x4 WMMA frags). Block = 8 waves arranged 4(rows) x 2(cols)
// covering a 256x128 super-tile so waves share A rows / B columns in WGP$ and L2.
// Requires M % 256 == 0, N % 128 == 0, K % 32 == 0.
enum { MODE_RELU_BF16 = 0, MODE_BIAS_F32 = 1, MODE_COS_F32 = 2 };

template <int MODE>
__global__ __launch_bounds__(256)
void k_gemm_wmma(const __bf16* __restrict__ A,
                 const __bf16* __restrict__ BT,
                 const float*  __restrict__ bias,
                 void* __restrict__ outp,
                 int M, int N, int K) {
  const int lane   = threadIdx.x & 31;
  const int wave   = threadIdx.x >> 5;
  const int superN = N >> 7;                  // 128-col block super-tiles
  const int sm = blockIdx.x / superN;
  const int sn = blockIdx.x - sm * superN;
  const int wr = wave >> 1;                   // 0..3 : wave row within block
  const int wc = wave & 1;                    // 0..1 : wave col within block
  const int r0 = (sm * 4 + wr) << 6;          // 64-row wave tile origin
  const int c0 = (sn * 2 + wc) << 6;          // 64-col wave tile origin
  const int lr = lane & 15;                   // row(A) / col(B,C,D) within 16
  const int lh = lane >> 4;                   // half-wave selector

  // Per-lane fragment base pointers (16 contiguous K per lane from lh*16, same k order for A and B)
  const __bf16* aptr[4];
#pragma unroll
  for (int i = 0; i < 4; ++i)
    aptr[i] = A + (size_t)(r0 + i * 16 + lr) * K + lh * 16;
  const __bf16* bptr[4];
#pragma unroll
  for (int j = 0; j < 4; ++j)
    bptr[j] = BT + (size_t)(c0 + j * 16 + lr) * K + lh * 16;

  f32x8 acc[4][4];
#pragma unroll
  for (int i = 0; i < 4; ++i)
#pragma unroll
    for (int j = 0; j < 4; ++j) acc[i][j] = (f32x8)0.0f;

  for (int kk = 0; kk < K; kk += 32) {
    // B fragments: 16 contiguous K values per lane (column c0 + j*16 + lr)
    bf16x16 bfr[4];
#pragma unroll
    for (int j = 0; j < 4; ++j) bfr[j] = *(const bf16x16*)(bptr[j] + kk);

    // A fragments: elems 0..15 at K = kk + lh*16 + e (contiguous, matches B)
    bf16x16 afr[4];
#pragma unroll
    for (int i = 0; i < 4; ++i) {
      bf16x8 lo = *(const bf16x8*)(aptr[i] + kk);
      bf16x8 hi = *(const bf16x8*)(aptr[i] + kk + 8);
      afr[i] = __builtin_shufflevector(lo, hi,
                                       0,1,2,3,4,5,6,7,8,9,10,11,12,13,14,15);
    }

#pragma unroll
    for (int i = 0; i < 4; ++i)
#pragma unroll
      for (int j = 0; j < 4; ++j)
        acc[i][j] = __builtin_amdgcn_wmma_f32_16x16x32_bf16(
            false, afr[i], false, bfr[j], (short)0, acc[i][j], false, false);
  }

  // Epilogue. C/D layout: VGPR e -> row r0 + i*16 + lh*8 + e, col c0 + j*16 + lr.
#pragma unroll
  for (int j = 0; j < 4; ++j) {
    const int col = c0 + j * 16 + lr;
    float bv = 0.0f;
    if constexpr (MODE != MODE_COS_F32) bv = bias[col];
#pragma unroll
    for (int i = 0; i < 4; ++i) {
#pragma unroll
      for (int e = 0; e < 8; ++e) {
        const size_t row = (size_t)(r0 + i * 16 + lh * 8 + e);
        float v = acc[i][j][e];
        if constexpr (MODE == MODE_RELU_BF16) {
          v = fmaxf(v + bv, 0.0f);
          ((__bf16*)outp)[row * N + col] = (__bf16)v;
        } else if constexpr (MODE == MODE_BIAS_F32) {
          ((float*)outp)[row * N + col] = v + bv;
        } else {
          ((float*)outp)[row * N + col] = (1.0f + v) * 0.5f;
        }
      }
    }
  }
}

// ---------------- row L2-normalize fp32 -> bf16 ----------------
__global__ __launch_bounds__(256)
void k_rownorm_bf16(const float* __restrict__ in, __bf16* __restrict__ out, int E) {
  __shared__ float red[256];
  const int row = blockIdx.x, tid = threadIdx.x;
  const float* r = in + (size_t)row * E;
  float s = 0.0f;
  for (int i = tid; i < E; i += 256) { float v = r[i]; s += v * v; }
  red[tid] = s;
  __syncthreads();
  for (int off = 128; off > 0; off >>= 1) {
    if (tid < off) red[tid] += red[tid + off];
    __syncthreads();
  }
  const float inv = 1.0f / fmaxf(sqrtf(red[0]), EPSN);
  for (int i = tid; i < E; i += 256)
    out[(size_t)row * E + i] = (__bf16)(r[i] * inv);
}

// ---------------- value head: sigmoid(logits_row . vw + vb) ----------------
__global__ __launch_bounds__(256)
void k_value(const float* __restrict__ logits, const float* __restrict__ vw,
             const float* __restrict__ vb, float* __restrict__ out, int Nn) {
  __shared__ float red[256];
  const int row = blockIdx.x, tid = threadIdx.x;
  const float* r = logits + (size_t)row * Nn;
  float s = 0.0f;
  for (int i = tid; i < Nn; i += 256) s += r[i] * vw[i];
  red[tid] = s;
  __syncthreads();
  for (int off = 128; off > 0; off >>= 1) {
    if (tid < off) red[tid] += red[tid + off];
    __syncthreads();
  }
  if (tid == 0) {
    const float x = red[0] + vb[0];
    out[row] = 1.0f / (1.0f + expf(-x));
  }
}

extern "C" void kernel_launch(void* const* d_in, const int* in_sizes, int n_in,
                              void* d_out, int out_size, void* d_ws, size_t ws_size,
                              hipStream_t stream) {
  (void)in_sizes; (void)n_in; (void)out_size; (void)ws_size;
  constexpr int B = 2048, Nsp = 8192, D = 1024, H = 4096, E = 1024;

  const float* msg = (const float*)d_in[0];
  const float* spk = (const float*)d_in[1];
  const float* mw1 = (const float*)d_in[2];
  const float* mb1 = (const float*)d_in[3];
  const float* mw2 = (const float*)d_in[4];
  const float* mb2 = (const float*)d_in[5];
  const float* cw1 = (const float*)d_in[6];
  const float* cb1 = (const float*)d_in[7];
  const float* cw2 = (const float*)d_in[8];
  const float* cb2 = (const float*)d_in[9];
  const float* vw  = (const float*)d_in[10];
  const float* vb  = (const float*)d_in[11];

  float* logits = (float*)d_out;                 // [B, Nsp]
  float* value  = logits + (size_t)B * Nsp;      // [B]

  char* ws = (char*)d_ws;
  const size_t MiB = 1u << 20;
  __bf16* Xm   = (__bf16*)(ws);              // 4  MiB  [B,D]    (reused as Mn)
  __bf16* Xc   = (__bf16*)(ws + 4 * MiB);    // 16 MiB  [Nsp,D]  (reused as Cn)
  __bf16* W1T  = (__bf16*)(ws + 20 * MiB);   // 8  MiB  [H,D]
  __bf16* W2T  = (__bf16*)(ws + 28 * MiB);   // 8  MiB  [E,H]
  __bf16* Hbuf = (__bf16*)(ws + 36 * MiB);   // 64 MiB  [Nsp,H] max
  float*  Act2 = (float*) (ws + 100 * MiB);  // 32 MiB  [Nsp,E] max

  auto cdiv = [](int a, int b) { return (a + b - 1) / b; };
  auto gemm_blocks = [](int M, int N) { return (M / 256) * (N / 128); };

  // --- message path ---
  k_f32_to_bf16<<<cdiv(B * D, 256), 256, 0, stream>>>(msg, Xm, B * D);
  k_transpose_bf16<<<dim3(H / 32, D / 32), dim3(32, 8), 0, stream>>>(mw1, W1T, D, H);
  k_transpose_bf16<<<dim3(E / 32, H / 32), dim3(32, 8), 0, stream>>>(mw2, W2T, H, E);
  k_gemm_wmma<MODE_RELU_BF16><<<gemm_blocks(B, H), 256, 0, stream>>>(
      Xm, W1T, mb1, Hbuf, B, H, D);
  k_gemm_wmma<MODE_BIAS_F32><<<gemm_blocks(B, E), 256, 0, stream>>>(
      Hbuf, W2T, mb2, Act2, B, E, H);
  k_rownorm_bf16<<<B, 256, 0, stream>>>(Act2, Xm /* Mn */, E);

  // --- speaker (context) path ---
  k_f32_to_bf16<<<cdiv(Nsp * D, 256), 256, 0, stream>>>(spk, Xc, Nsp * D);
  k_transpose_bf16<<<dim3(H / 32, D / 32), dim3(32, 8), 0, stream>>>(cw1, W1T, D, H);
  k_transpose_bf16<<<dim3(E / 32, H / 32), dim3(32, 8), 0, stream>>>(cw2, W2T, H, E);
  k_gemm_wmma<MODE_RELU_BF16><<<gemm_blocks(Nsp, H), 256, 0, stream>>>(
      Xc, W1T, cb1, Hbuf, Nsp, H, D);
  k_gemm_wmma<MODE_BIAS_F32><<<gemm_blocks(Nsp, E), 256, 0, stream>>>(
      Hbuf, W2T, cb2, Act2, Nsp, E, H);
  k_rownorm_bf16<<<Nsp, 256, 0, stream>>>(Act2, Xc /* Cn */, E);

  // --- cosine GEMM: logits = (1 + Mn @ Cn^T) / 2 ; Cn row-major IS Cn^T's BT ---
  k_gemm_wmma<MODE_COS_F32><<<gemm_blocks(B, Nsp), 256, 0, stream>>>(
      Xm, Xc, nullptr, logits, B, Nsp, E);

  // --- value head ---
  k_value<<<B, 256, 0, stream>>>(logits, vw, vb, value, Nsp);
}
